// GAT_70531952934941
// MI455X (gfx1250) — matrix operand order repaired and checked by reading, not compiled, not run on hardware
//
#include <hip/hip_runtime.h>

// ---------------------------------------------------------------------------
// GATv2 x2 layers for MI455X (gfx1250, wave32).
// GEMMs: operands pre-converted to fragment-friendly f16 (A row-major,
// B transposed N x K), fragments loaded straight from global (L2-resident)
// as b128s, 4x v_wmma_f32_16x16x32_f16 per wave per K-step, no LDS/barriers.
// Edge softmax/aggregation via hardware atomics (memory-bound phase).
// ---------------------------------------------------------------------------

typedef __attribute__((ext_vector_type(16))) _Float16 v16h;
typedef __attribute__((ext_vector_type(8)))  _Float16 v8h;
typedef __attribute__((ext_vector_type(8)))  float    v8f;

// ---------------- f32 -> f16 row-major copy (n multiple of 8) -------------
__global__ __launch_bounds__(256) void convert_f16_kernel(
    const float* __restrict__ in, _Float16* __restrict__ out, long long n8)
{
    const long long i = (long long)blockIdx.x * blockDim.x + threadIdx.x;
    if (i >= n8) return;
    const float4* p = (const float4*)(in + i * 8);
    const float4 a = p[0], b = p[1];
    v8h h = { (_Float16)a.x, (_Float16)a.y, (_Float16)a.z, (_Float16)a.w,
              (_Float16)b.x, (_Float16)b.y, (_Float16)b.z, (_Float16)b.w };
    *(v8h*)(out + i * 8) = h;
}

// ---------------- W (K x N f32) -> Bt (N x K f16) transposed --------------
__global__ __launch_bounds__(256) void transpose_f16_kernel(
    const float* __restrict__ B, _Float16* __restrict__ Bt, int K, int N)
{
    const long long i = (long long)blockIdx.x * blockDim.x + threadIdx.x;
    if (i >= (long long)K * N) return;
    const int n = (int)(i / K);
    const int k = (int)(i % K);
    Bt[(size_t)n * K + k] = (_Float16)B[(size_t)k * N + n];
}

// ---------------- WMMA GEMM: C[M,N] = Ah[M,K] * Bt[N,K]^T -----------------
// block: 256 threads = 8 waves (2 x 4); block tile 32 x 256; wave 16 x 64.
__global__ __launch_bounds__(256) void gemm_wmma_f16(
    const _Float16* __restrict__ Ah,   // M x K row-major f16
    const _Float16* __restrict__ Bt,   // N x K row-major f16 (B transposed)
    float* __restrict__ C, int M, int N, int K)
{
    const int tid  = threadIdx.x;
    const int lane = tid & 31;
    const int wave = tid >> 5;
    const int wrow = wave & 1;       // 0..1  (16 rows each)
    const int wcol = wave >> 1;      // 0..3  (64 cols each)
    const int hi16 = lane >> 4;      // half-wave select
    const int l16  = lane & 15;

    int arow = blockIdx.x * 32 + wrow * 16 + l16;
    if (arow >= M) arow = M - 1;                       // safety (M%32==0 here)
    const int colBase = blockIdx.y * 256 + wcol * 64;

    // A fragment source: row arow, contiguous 32B run at K hi16*16
    const _Float16* ap = Ah + (size_t)arow * K;
    // B fragment sources: 4 column tiles; contiguous 32B run at K hi16*16
    const _Float16* bp0 = Bt + (size_t)(colBase +  0 + l16) * K + hi16 * 16;
    const _Float16* bp1 = Bt + (size_t)(colBase + 16 + l16) * K + hi16 * 16;
    const _Float16* bp2 = Bt + (size_t)(colBase + 32 + l16) * K + hi16 * 16;
    const _Float16* bp3 = Bt + (size_t)(colBase + 48 + l16) * K + hi16 * 16;

    v8f acc0 = {}, acc1 = {}, acc2 = {}, acc3 = {};

    for (int k0 = 0; k0 < K; k0 += 32) {
        __builtin_prefetch(ap + k0 + 64, 0, 0);        // global_prefetch_b8
        const v8h a0 = *(const v8h*)(ap + k0 + hi16 * 16);
        const v8h a1 = *(const v8h*)(ap + k0 + hi16 * 16 + 8);
        const v16h af = __builtin_shufflevector(a0, a1,
                          0,1,2,3,4,5,6,7,8,9,10,11,12,13,14,15);

        const v8h b00 = *(const v8h*)(bp0 + k0);
        const v8h b01 = *(const v8h*)(bp0 + k0 + 8);
        const v16h bf0 = __builtin_shufflevector(b00, b01,
                          0,1,2,3,4,5,6,7,8,9,10,11,12,13,14,15);
        acc0 = __builtin_amdgcn_wmma_f32_16x16x32_f16(
                   false, af, false, bf0, (short)0, acc0, false, false);

        const v8h b10 = *(const v8h*)(bp1 + k0);
        const v8h b11 = *(const v8h*)(bp1 + k0 + 8);
        const v16h bf1 = __builtin_shufflevector(b10, b11,
                          0,1,2,3,4,5,6,7,8,9,10,11,12,13,14,15);
        acc1 = __builtin_amdgcn_wmma_f32_16x16x32_f16(
                   false, af, false, bf1, (short)0, acc1, false, false);

        const v8h b20 = *(const v8h*)(bp2 + k0);
        const v8h b21 = *(const v8h*)(bp2 + k0 + 8);
        const v16h bf2 = __builtin_shufflevector(b20, b21,
                          0,1,2,3,4,5,6,7,8,9,10,11,12,13,14,15);
        acc2 = __builtin_amdgcn_wmma_f32_16x16x32_f16(
                   false, af, false, bf2, (short)0, acc2, false, false);

        const v8h b30 = *(const v8h*)(bp3 + k0);
        const v8h b31 = *(const v8h*)(bp3 + k0 + 8);
        const v16h bf3 = __builtin_shufflevector(b30, b31,
                          0,1,2,3,4,5,6,7,8,9,10,11,12,13,14,15);
        acc3 = __builtin_amdgcn_wmma_f32_16x16x32_f16(
                   false, af, false, bf3, (short)0, acc3, false, false);
    }

    // D layout: lane holds column l16; VGPR i -> row i + hi16*8
    const int row0 = blockIdx.x * 32 + wrow * 16 + hi16 * 8;
    const v8f accs[4] = { acc0, acc1, acc2, acc3 };
    #pragma unroll
    for (int j = 0; j < 4; ++j) {
        const int col = colBase + j * 16 + l16;
        #pragma unroll
        for (int i = 0; i < 8; ++i) {
            const int r = row0 + i;
            if (r < M) C[(size_t)r * N + col] = accs[j][i];
        }
    }
}

// ---------------- helpers ----------------
__device__ __forceinline__ unsigned fkey(float x) {       // monotone f32->u32
    unsigned u = __float_as_uint(x);
    return (u & 0x80000000u) ? ~u : (u | 0x80000000u);
}
__device__ __forceinline__ float unkey(unsigned k) {
    unsigned u = (k & 0x80000000u) ? (k ^ 0x80000000u) : ~k;
    return __uint_as_float(u);
}
__device__ __forceinline__ void edge_sd(const int* __restrict__ ei, int e, int E,
                                        int& s, int& d) {
    if (e < E) { s = ei[e]; d = ei[E + e]; } else { s = e - E; d = e - E; }
}

__global__ void fill_u32_kernel(unsigned* __restrict__ p, unsigned v, long long n) {
    long long i = (long long)blockIdx.x * blockDim.x + threadIdx.x;
    if (i < n) p[i] = v;
}
__global__ void init_bias_kernel(float* __restrict__ out, const float* __restrict__ bias,
                                 long long total, int C) {
    long long i = (long long)blockIdx.x * blockDim.x + threadIdx.x;
    if (i < total) out[i] = bias[(int)(i % C)];
}

// ---------------- edge pass 1: logits + segment max ----------------
__global__ __launch_bounds__(256) void edge_logits_kernel(
    const float* __restrict__ xl, const float* __restrict__ xr,
    const float* __restrict__ att, const int* __restrict__ ei,
    int E, int Nn, float* __restrict__ logits, unsigned* __restrict__ nodeMaxU)
{
    const long long idx = (long long)blockIdx.x * blockDim.x + threadIdx.x;
    const long long Et  = (long long)E + Nn;
    if (idx >= Et * 8) return;
    const int e = (int)(idx >> 3);
    const int h = (int)(idx & 7);
    int s, d; edge_sd(ei, e, E, s, d);

    const float4* lp = (const float4*)(xl + (size_t)s * 512 + h * 64);
    const float4* rp = (const float4*)(xr + (size_t)d * 512 + h * 64);
    const float4* ap = (const float4*)(att + h * 64);
    float acc = 0.f;
    #pragma unroll
    for (int i = 0; i < 16; ++i) {
        const float4 a = lp[i], b = rp[i], w = ap[i];
        float v;
        v = a.x + b.x; v = v > 0.f ? v : 0.2f * v; acc += v * w.x;
        v = a.y + b.y; v = v > 0.f ? v : 0.2f * v; acc += v * w.y;
        v = a.z + b.z; v = v > 0.f ? v : 0.2f * v; acc += v * w.z;
        v = a.w + b.w; v = v > 0.f ? v : 0.2f * v; acc += v * w.w;
    }
    logits[idx] = acc;
    atomicMax(&nodeMaxU[(size_t)d * 8 + h], fkey(acc));
}

// ---------------- edge pass 2: p = exp(l - max[dst]); segment sum ----------
__global__ __launch_bounds__(256) void edge_exp_sum_kernel(
    float* __restrict__ lp, const unsigned* __restrict__ maxU,
    float* __restrict__ sum, const int* __restrict__ ei, int E, int Nn)
{
    const long long idx = (long long)blockIdx.x * blockDim.x + threadIdx.x;
    const long long Et  = (long long)E + Nn;
    if (idx >= Et * 8) return;
    const int e = (int)(idx >> 3);
    const int h = (int)(idx & 7);
    int s, d; edge_sd(ei, e, E, s, d);
    const float m = unkey(maxU[(size_t)d * 8 + h]);
    const float p = __expf(lp[idx] - m);
    lp[idx] = p;
    atomicAdd(&sum[(size_t)d * 8 + h], p);
}

// ---------------- layer-1 aggregation (concat heads) ----------------
__global__ __launch_bounds__(256) void edge_agg_concat_kernel(
    const float* __restrict__ xl, const float* __restrict__ p,
    const float* __restrict__ sum, const int* __restrict__ ei,
    int E, int Nn, float* __restrict__ accum)
{
    const long long idx = (long long)blockIdx.x * blockDim.x + threadIdx.x;
    const long long Et  = (long long)E + Nn;
    if (idx >= Et * 512) return;
    const int e = (int)(idx >> 9);
    const int c = (int)(idx & 511);
    const int h = c >> 6;
    int s, d; edge_sd(ei, e, E, s, d);
    const float alpha = p[(size_t)e * 8 + h] / sum[(size_t)d * 8 + h];
    atomicAdd(&accum[(size_t)d * 512 + c], xl[(size_t)s * 512 + c] * alpha);
}

// ---------------- elementwise elu ----------------
__global__ void elu_kernel(float* __restrict__ h, long long n) {
    long long i = (long long)blockIdx.x * blockDim.x + threadIdx.x;
    if (i < n) { const float v = h[i]; h[i] = v > 0.f ? v : (__expf(v) - 1.f); }
}

// ---------------- layer-2 aggregation (mean over heads) ----------------
__global__ __launch_bounds__(256) void edge_agg_mean_kernel(
    const float* __restrict__ xl, const float* __restrict__ p,
    const float* __restrict__ sum, const int* __restrict__ ei,
    int E, int Nn, float* __restrict__ out)
{
    const long long idx = (long long)blockIdx.x * blockDim.x + threadIdx.x;
    const long long Et  = (long long)E + Nn;
    if (idx >= Et * 64) return;
    const int e = (int)(idx >> 6);
    const int c = (int)(idx & 63);
    int s, d; edge_sd(ei, e, E, s, d);
    float v = 0.f;
    #pragma unroll
    for (int h = 0; h < 8; ++h) {
        const float alpha = p[(size_t)e * 8 + h] / sum[(size_t)d * 8 + h];
        v += xl[(size_t)s * 512 + h * 64 + c] * alpha;
    }
    atomicAdd(&out[(size_t)d * 64 + c], 0.125f * v);
}

// ---------------------------------------------------------------------------
extern "C" void kernel_launch(void* const* d_in, const int* in_sizes, int n_in,
                              void* d_out, int out_size, void* d_ws, size_t ws_size,
                              hipStream_t stream)
{
    const float* x    = (const float*)d_in[0];
    const int*   ei   = (const int*)  d_in[1];
    const float* Wl1  = (const float*)d_in[2];
    const float* Wr1  = (const float*)d_in[3];
    const float* att1 = (const float*)d_in[4];
    const float* b1   = (const float*)d_in[5];
    const float* Wl2  = (const float*)d_in[6];
    const float* Wr2  = (const float*)d_in[7];
    const float* att2 = (const float*)d_in[8];
    const float* b2   = (const float*)d_in[9];
    float* out = (float*)d_out;

    const int Nn = in_sizes[0] / 64;   // 20000 nodes
    const int E  = in_sizes[1] / 2;    // 320000 edges
    const long long Et = (long long)E + Nn;
    const int D = 64, HD = 512;

    // workspace carve-up
    float* ws = (float*)d_ws;
    const size_t nHD = (size_t)Nn * HD;
    float*    bufA    = ws;                                  // xl  (N x 512) f32
    float*    bufB    = bufA + nHD;                          // xr  (N x 512) f32
    float*    bufH    = bufB + nHD;                          // h   (N x 512) f32
    float*    bufL    = bufH + nHD;                          // edge p (Et x 8) f32
    unsigned* nodeMax = (unsigned*)(bufL + (size_t)Et * 8);  // N x 8 u32
    float*    nodeSum = (float*)(nodeMax + (size_t)Nn * 8);  // N x 8 f32
    _Float16* Ah      = (_Float16*)(nodeSum + (size_t)Nn * 8); // N x 512 f16 max
    _Float16* Bt      = Ah + nHD;                            // 512 x 512 f16 max

    const dim3 blk(256);
    const dim3 gemmGrid((Nn + 31) / 32, HD / 256);
    auto nblk = [](long long t) { return (unsigned)((t + 255) / 256); };

    // ================= layer 1 =================
    convert_f16_kernel<<<nblk((long long)Nn * D / 8), blk, 0, stream>>>(
        x, Ah, (long long)Nn * D / 8);

    transpose_f16_kernel<<<nblk((long long)D * HD), blk, 0, stream>>>(Wl1, Bt, D, HD);
    gemm_wmma_f16<<<gemmGrid, blk, 0, stream>>>(Ah, Bt, bufA, Nn, HD, D);
    transpose_f16_kernel<<<nblk((long long)D * HD), blk, 0, stream>>>(Wr1, Bt, D, HD);
    gemm_wmma_f16<<<gemmGrid, blk, 0, stream>>>(Ah, Bt, bufB, Nn, HD, D);

    fill_u32_kernel<<<nblk((long long)Nn * 16), blk, 0, stream>>>(nodeMax, 0u, (long long)Nn * 16);
    init_bias_kernel<<<nblk((long long)nHD), blk, 0, stream>>>(bufH, b1, (long long)nHD, HD);

    edge_logits_kernel<<<nblk(Et * 8), blk, 0, stream>>>(bufA, bufB, att1, ei, E, Nn, bufL, nodeMax);
    edge_exp_sum_kernel<<<nblk(Et * 8), blk, 0, stream>>>(bufL, nodeMax, nodeSum, ei, E, Nn);
    edge_agg_concat_kernel<<<nblk(Et * 512), blk, 0, stream>>>(bufA, bufL, nodeSum, ei, E, Nn, bufH);
    elu_kernel<<<nblk((long long)nHD), blk, 0, stream>>>(bufH, (long long)nHD);

    // ================= layer 2 =================
    convert_f16_kernel<<<nblk((long long)nHD / 8), blk, 0, stream>>>(
        bufH, Ah, (long long)nHD / 8);

    transpose_f16_kernel<<<nblk((long long)HD * HD), blk, 0, stream>>>(Wl2, Bt, HD, HD);
    gemm_wmma_f16<<<gemmGrid, blk, 0, stream>>>(Ah, Bt, bufA, Nn, HD, HD);
    transpose_f16_kernel<<<nblk((long long)HD * HD), blk, 0, stream>>>(Wr2, Bt, HD, HD);
    gemm_wmma_f16<<<gemmGrid, blk, 0, stream>>>(Ah, Bt, bufB, Nn, HD, HD);

    fill_u32_kernel<<<nblk((long long)Nn * 16), blk, 0, stream>>>(nodeMax, 0u, (long long)Nn * 16);
    init_bias_kernel<<<nblk((long long)Nn * 64), blk, 0, stream>>>(out, b2, (long long)Nn * 64, 64);

    edge_logits_kernel<<<nblk(Et * 8), blk, 0, stream>>>(bufA, bufB, att2, ei, E, Nn, bufL, nodeMax);
    edge_exp_sum_kernel<<<nblk(Et * 8), blk, 0, stream>>>(bufL, nodeMax, nodeSum, ei, E, Nn);
    edge_agg_mean_kernel<<<nblk(Et * 64), blk, 0, stream>>>(bufA, bufL, nodeSum, ei, E, Nn, out);

    (void)n_in; (void)out_size; (void)ws_size;
}
